// LB_CNN_Gibbs_34840774705393
// MI455X (gfx1250) — matrix operand order repaired and checked by reading, not compiled, not run on hardware
//
#include <hip/hip_runtime.h>
#include <hip/hip_bf16.h>
#include <stdint.h>

// Problem dims (fixed by the reference)
#define N_PTS 2048
#define F_DIM 1024
#define M_DIM 128
#define C_CLS 100
#define S_SMP 32
#define NPB   4     // datapoints (waves) per Gibbs block (128 threads)

typedef __attribute__((ext_vector_type(16))) __bf16 v16bf;
typedef __attribute__((ext_vector_type(8)))  __bf16 v8bf;
typedef __attribute__((ext_vector_type(8)))  float  v8f;
typedef __attribute__((ext_vector_type(4)))  unsigned int u32x4;
typedef __attribute__((ext_vector_type(8)))  int i32x8;
typedef __attribute__((ext_vector_type(4)))  int i32x4;

__device__ __forceinline__ uint16_t f2bfbits(float f) {
  uint32_t u = __builtin_bit_cast(uint32_t, f);
  uint32_t r = (u + 0x7FFFu + ((u >> 16) & 1u)) >> 16;   // round-to-nearest-even
  return (uint16_t)r;
}

__device__ __forceinline__ v16bf combine16(v8bf lo, v8bf hi) {
  return __builtin_shufflevector(lo, hi, 0,1,2,3,4,5,6,7,8,9,10,11,12,13,14,15);
}

// Raw LDS 128-bit loads at byte addresses, as volatile inline asm: the TDM
// writes LDS behind the compiler's back, so ordinary C++ loads of the staging
// buffer get folded to undef (observed rounds 2-3).
// lds_b128: embeds its own wait (used in the non-TDM fallback).
// lds_b128_async: no wait — pair with the counted wait helpers below, which
// take the landed registers as "+v" so consumers cannot be scheduled earlier.
__device__ __forceinline__ v8bf lds_b128(uint32_t byte_addr) {
  v8bf r;
  asm volatile("ds_load_b128 %0, %1\n\ts_wait_dscnt 0x0"
               : "=v"(r) : "v"(byte_addr));
  return r;
}
__device__ __forceinline__ v8bf lds_b128_async(uint32_t byte_addr) {
  v8bf r;
  asm volatile("ds_load_b128 %0, %1" : "=v"(r) : "v"(byte_addr));
  return r;
}
// DS returns are in-order: wait until <=2 outstanding -> everything except the
// most recently issued pair has landed.
__device__ __forceinline__ void wait_ds2_4(v8bf& a, v8bf& b, v8bf& c, v8bf& d) {
  asm volatile("s_wait_dscnt 0x2" : "+v"(a), "+v"(b), "+v"(c), "+v"(d));
}
__device__ __forceinline__ void wait_ds2_2(v8bf& a, v8bf& b) {
  asm volatile("s_wait_dscnt 0x2" : "+v"(a), "+v"(b));
}
__device__ __forceinline__ void wait_ds0_2(v8bf& a, v8bf& b) {
  asm volatile("s_wait_dscnt 0x0" : "+v"(a), "+v"(b));
}

// Stateless counter-based RNG (Bernoulli draws; cannot match JAX threefry anyway)
__device__ __forceinline__ float rnd01(uint32_t a, uint32_t b) {
  uint32_t h = a * 0x9E3779B1u + b * 0x85EBCA77u;
  h ^= h >> 16; h *= 0x7FEB352Du;
  h ^= h >> 15; h *= 0x846CA68Bu;
  h ^= h >> 16;
  return (float)(h >> 8) * (1.0f / 16777216.0f);
}

// ---------------------------------------------------------------------------
// Tensor Data Mover: 2D tile (tile_d0 x tile_d1 elements, 2-byte data) from
// global row-major tensor (row stride = stride0 elements) into LDS at lds_off.
// D# packing per CDNA5 ISA.
// ---------------------------------------------------------------------------
#if __has_builtin(__builtin_amdgcn_tensor_load_to_lds)
#define HAVE_TDM 1
__device__ __forceinline__ void tdm_load_2d(uint32_t lds_off, const void* gaddr,
                                            uint32_t tile_d0, uint32_t tile_d1,
                                            uint32_t tens_d0, uint32_t tens_d1,
                                            uint32_t stride0) {
  uint64_t ga = (uint64_t)(uintptr_t)gaddr;
  u32x4 g0;
  g0[0] = 1u;                                        // count=1 (valid), user mode
  g0[1] = lds_off;                                   // lds_addr (bytes)
  g0[2] = (uint32_t)(ga & 0xFFFFFFFFu);              // global_addr[31:0]
  g0[3] = (uint32_t)((ga >> 32) & 0x01FFFFFFu)       // global_addr[56:32]
        | (2u << 30);                                // type = 2 ("image")
  i32x8 g1;
  g1[0] = (int)(1u << 16);                           // data_size=1 (2B); wg_mask=0
  g1[1] = (int)((tens_d0 & 0xFFFFu) << 16);          // tensor_dim0[15:0] @ bits 63:48
  g1[2] = (int)((tens_d0 >> 16) | ((tens_d1 & 0xFFFFu) << 16)); // dim0 hi | dim1 lo
  g1[3] = (int)((tens_d1 >> 16) | ((tile_d0 & 0xFFFFu) << 16)); // dim1 hi | tile_dim0
  g1[4] = (int)(tile_d1 & 0xFFFFu);                  // tile_dim1; tile_dim2=0
  g1[5] = (int)stride0;                              // tensor_dim0_stride[31:0]
  g1[6] = 0;                                         // stride0[47:32] | stride1[15:0]
  g1[7] = 0;                                         // stride1[47:16]
  i32x4 gz = {0, 0, 0, 0};
#if __clang_major__ >= 23
  i32x8 gz8 = {0, 0, 0, 0, 0, 0, 0, 0};
  __builtin_amdgcn_tensor_load_to_lds(g0, g1, gz, gz, gz8, 0);
#else
  __builtin_amdgcn_tensor_load_to_lds(g0, g1, gz, gz, 0);
#endif
}
#else
#define HAVE_TDM 0
#endif

__global__ void k_zero(float* out) { out[0] = 0.0f; }

// f32 -> bf16 bulk convert (4 elements / thread), one bandwidth pass
__global__ void k_cvt(const float* __restrict__ src, uint16_t* __restrict__ dst, int n4) {
  int i = blockIdx.x * blockDim.x + threadIdx.x;
  if (i >= n4) return;
  float4 v = ((const float4*)src)[i];
  ushort4 o;
  o.x = f2bfbits(v.x); o.y = f2bfbits(v.y);
  o.z = f2bfbits(v.z); o.w = f2bfbits(v.w);
  ((ushort4*)dst)[i] = o;
}

// tab+/-[j*C + c] = exp(+-2 * beta[c][j])
__global__ void k_tabs(const float* __restrict__ beta,
                       float* __restrict__ tabp, float* __restrict__ tabn) {
  int t = blockIdx.x * blockDim.x + threadIdx.x;
  if (t >= M_DIM * C_CLS) return;
  int j = t / C_CLS, c = t % C_CLS;
  float b = beta[(size_t)c * M_DIM + j];
  tabp[t] = __expf( 2.0f * b);
  tabn[t] = __expf(-2.0f * b);
}

// linear_pi = alpha0 + x @ alpha^T  (2048x128x1024), bf16 WMMA.
// Block = 8 waves = 128 output rows x all 128 cols. Double-buffered TDM
// pipeline: wave 0 issues the next K-chunk's 128x32 bf16 tiles of x and alpha
// into LDS buffer cur^1 while all waves run WMMAs from buffer cur. Fragment
// reads keep a 2-deep ds_load_b128 pipeline with counted dscnt waits.
// LDS map (stage is the only shared object -> LDS offset 0, same convention
// used by the TDM descriptors): buf b at b*16384; x tile at +0, alpha at +8192.
__global__ void __launch_bounds__(256)
k_linear_pi(const uint16_t* __restrict__ xbf, const uint16_t* __restrict__ abf,
            const float* __restrict__ alpha0,
            float* __restrict__ linpi, float* __restrict__ pipos) {
  __shared__ __align__(16) uint16_t stage[2][2 * 128 * 32];   // 2 x 16 KB
  const int lane = threadIdx.x & 31;
  const int wave = threadIdx.x >> 5;
  const int lrow = lane & 15;
  const int hi   = lane >> 4;
  const int rbb  = blockIdx.x * 128;    // block output-row base
  const uint16_t* xrow = xbf + (size_t)rbb * F_DIM;

  // Per-lane byte addresses inside a buffer (row = 32 bf16 = 64 B)
  const uint32_t ax_base = (uint32_t)((wave * 16 + lrow) * 64 + hi * 32);
  const uint32_t ab_base = (uint32_t)(8192 + lrow * 64 + hi * 32);

  v8f acc[8] = {};

#if HAVE_TDM
  // Prologue: fill buffer 0 with the kb=0 tiles.
  if (wave == 0) {
    tdm_load_2d(0u,    xrow, 32u, 128u, F_DIM, 128u, F_DIM);
    tdm_load_2d(8192u, abf,  32u, 128u, F_DIM, 128u, F_DIM);
    __builtin_amdgcn_s_wait_tensorcnt(0);
  }
  __syncthreads();

  int cur = 0;
  for (int kb = 0; kb < F_DIM; kb += 32, cur ^= 1) {
    // Kick off the next chunk's DMA into the other buffer.
    if (wave == 0 && kb + 32 < F_DIM) {
      uint32_t nb = (uint32_t)(cur ^ 1) * 16384u;
      tdm_load_2d(nb,         xrow + kb + 32, 32u, 128u,
                  (uint32_t)(F_DIM - kb - 32), 128u, F_DIM);
      tdm_load_2d(nb + 8192u, abf + kb + 32,  32u, 128u,
                  (uint32_t)(F_DIM - kb - 32), 128u, F_DIM);
    }

    const uint32_t buf = (uint32_t)cur * 16384u;

    // A-frag (16x32): K = hi*16 + {0..15}; B pair 0 right behind.
    v8bf a0 = lds_b128_async(buf + ax_base);
    v8bf a1 = lds_b128_async(buf + ax_base + 16u);
    v8bf bl = lds_b128_async(buf + ab_base);
    v8bf bh = lds_b128_async(buf + ab_base + 16u);
    v16bf a;

    #pragma unroll
    for (int cbt = 0; cbt < 8; cbt++) {
      v8bf nbl, nbh;
      if (cbt < 7) {   // keep next B pair in flight
        uint32_t abn = buf + ab_base + (uint32_t)((cbt + 1) * 1024);
        nbl = lds_b128_async(abn);
        nbh = lds_b128_async(abn + 16u);
      }
      if (cbt == 0) {                  // a0,a1,bl,bh landed; next pair in flight
        wait_ds2_4(a0, a1, bl, bh);
        a = combine16(a0, a1);
      } else if (cbt < 7) {
        wait_ds2_2(bl, bh);
      } else {
        wait_ds0_2(bl, bh);
      }
      v16bf b = combine16(bl, bh);
      acc[cbt] = __builtin_amdgcn_wmma_f32_16x16x32_bf16(
          false, a, false, b, (short)0, acc[cbt], false, false);
      if (cbt < 7) { bl = nbl; bh = nbh; }
    }

    if (wave == 0) __builtin_amdgcn_s_wait_tensorcnt(0);  // next buffer landed
    __syncthreads();  // everyone done reading cur; next buffer published
  }
#else
  for (int kb = 0; kb < F_DIM; kb += 32) {
    {  // cooperative fallback copy: 16 KB per K-step into buffer 0
      const uint32_t* gx = (const uint32_t*)(xrow + kb);
      const uint32_t* ga = (const uint32_t*)(abf + kb);
      uint32_t* sxw = (uint32_t*)&stage[0][0];
      uint32_t* saw = (uint32_t*)&stage[0][4096];
      for (int t = threadIdx.x; t < 128 * 16; t += 256) {
        int r = t >> 4, c2 = t & 15;   // 16 dwords (32 bf16) per row
        sxw[t] = gx[(size_t)r * (F_DIM / 2) + c2];
        saw[t] = ga[(size_t)r * (F_DIM / 2) + c2];
      }
    }
    __syncthreads();
    v16bf a = combine16(lds_b128(ax_base), lds_b128(ax_base + 16u));
    #pragma unroll
    for (int cbt = 0; cbt < 8; cbt++) {
      uint32_t ab = ab_base + (uint32_t)(cbt * 1024);
      v16bf b = combine16(lds_b128(ab), lds_b128(ab + 16u));
      acc[cbt] = __builtin_amdgcn_wmma_f32_16x16x32_bf16(
          false, a, false, b, (short)0, acc[cbt], false, false);
    }
    __syncthreads();
  }
#endif

  // C/D layout: VGPR v -> row rb + v + hi*8, lane&15 -> col
  const int rb = rbb + wave * 16;
  #pragma unroll
  for (int cbt = 0; cbt < 8; cbt++) {
    #pragma unroll
    for (int v = 0; v < 8; v++) {
      int row = rb + v + hi * 8;
      int col = cbt * 16 + lrow;
      float val = acc[cbt][v] + alpha0[col];
      linpi[(size_t)row * M_DIM + col] = val;
      pipos[(size_t)row * M_DIM + col] = 1.0f / (1.0f + __expf(-val));
    }
  }
}

// Gibbs scan + all moment terms. One wave = the 32 chains of one datapoint n.
// Chain state: Z as 128 bits in 4 u32; E_c = exp(lin_c) in LDS; T = sum_c E_c.
__global__ void __launch_bounds__(128)
k_gibbs(const float* __restrict__ pipos, const float* __restrict__ linpi,
        const float* __restrict__ beta0, const float* __restrict__ beta,
        const int* __restrict__ y,
        const float* __restrict__ tabp, const float* __restrict__ tabn,
        float* __restrict__ out) {
  __shared__ float Es[C_CLS * 128];  // [c*128 + tid], 51.2 KB
  const int tid  = threadIdx.x;
  const int lane = tid & 31;                       // chain s
  const int n    = blockIdx.x * NPB + (tid >> 5);  // datapoint
  const int yn   = y[n];
  const uint32_t tseed = (uint32_t)(n * S_SMP + lane);
  const float* pin = pipos + (size_t)n * M_DIM;

  // --- init: Z0 ~ 2*Bern(pi_pos)-1, bit=1 means z=+1 ---
  uint32_t Zb[4];
  #pragma unroll
  for (int w = 0; w < 4; w++) {
    uint32_t bits = 0;
    for (int b2 = 0; b2 < 32; b2++) {
      int m = w * 32 + b2;
      float u = rnd01(tseed, (uint32_t)m);
      bits |= (u < pin[m] ? 1u : 0u) << b2;
    }
    Zb[w] = bits;
  }

  // --- E_c = exp(beta0[c] + sum_m z_m * beta[c][m]) ; T = sum_c E_c ---
  float T = 0.0f;
  for (int c = 0; c < C_CLS; c++) {
    float lin = beta0[c];
    const float* br = beta + (size_t)c * M_DIM;
    #pragma unroll
    for (int w = 0; w < 4; w++) {
      uint32_t bits = Zb[w];
      for (int b2 = 0; b2 < 32; b2++) {
        float bm = br[w * 32 + b2];
        lin += ((bits >> b2) & 1u) ? bm : -bm;
      }
    }
    float e = __expf(lin);
    Es[c * 128 + tid] = e;
    T += e;
  }

  // --- Gibbs sweep: p = 1/(1+rho), rho = (1-pi)/pi * exp(-2*b_yj) * Spos/Sneg ---
  // zj=+1: Spos=T, Sneg=dot(E,tabn); zj=-1: Sneg=T, Spos=dot(E,tabp).
  // On a flip: E_c *= tab (same table as the dot), T <- S (exactly the dot).
  #pragma unroll
  for (int w = 0; w < 4; w++) {
    uint32_t bits = Zb[w];
    for (int b2 = 0; b2 < 32; b2++) {
      const int j = w * 32 + b2;
      const uint32_t zbit = (bits >> b2) & 1u;
      const float* tb = zbit ? (tabn + j * C_CLS) : (tabp + j * C_CLS);
      if (j + 1 < M_DIM) __builtin_prefetch(tabp + (j + 1) * C_CLS, 0, 0);
      float S = 0.0f;
      for (int c = 0; c < C_CLS; c++) S += Es[c * 128 + tid] * tb[c];
      float piv   = pin[j];
      float tny   = tabn[j * C_CLS + yn];          // exp(-2*beta[yn][j])
      float ratio = zbit ? (T / S) : (S / T);
      float rho   = (1.0f - piv) / piv * tny * ratio;
      float p     = 1.0f / (1.0f + rho);
      float u     = rnd01(tseed, 128u + (uint32_t)j);
      uint32_t znew = (u < p) ? 1u : 0u;
      if (znew != zbit) {
        for (int c = 0; c < C_CLS; c++) Es[c * 128 + tid] *= tb[c];
        T = S;
        bits ^= (1u << b2);
      }
    }
    Zb[w] = bits;
  }

  // --- moments, folded to scalar: TermI - TermII + TermIII - TermIV ---
  float accl = -__logf(T) * (1.0f / (float)S_SMP);   // -TermII piece (logsumexp = log T)

  // EI/EII via wave32 ballots: pc[m] = #chains with z_m=+1.
  uint32_t pc[4];
  #pragma unroll
  for (int w = 0; w < 4; w++) {
    uint32_t bits = Zb[w];
    uint32_t mypc = 0;
    for (int b2 = 0; b2 < 32; b2++) {
      unsigned mask = __builtin_amdgcn_ballot_w32(((bits >> b2) & 1u) != 0);
      if (b2 == lane) mypc = (uint32_t)__popc(mask);
    }
    pc[w] = mypc;
  }
  const float inv_s = 1.0f / (float)S_SMP;
  #pragma unroll
  for (int q = 0; q < 4; q++) {
    int m = q * 32 + lane;                // each m counted exactly once per wave
    float pcf = (float)pc[q];
    float EI  = pcf * inv_s;              // mean (z+1)/2
    float EII = (2.0f * pcf - (float)S_SMP) * inv_s;  // mean z
    float lp  = linpi[(size_t)n * M_DIM + m];
    accl += EII * beta[(size_t)yn * M_DIM + m];                 // TermI
    accl += EI * lp;                                            // TermIII
    accl -= fmaxf(lp, 0.0f) + log1pf(__expf(-fabsf(lp)));       // -TermIV
  }
  if (lane == 0) accl += beta0[yn];       // TermI bias

  #pragma unroll
  for (int off = 16; off >= 1; off >>= 1) accl += __shfl_xor(accl, off, 32);
  if (lane == 0) atomicAdd(out, accl);
}

extern "C" void kernel_launch(void* const* d_in, const int* in_sizes, int n_in,
                              void* d_out, int out_size, void* d_ws, size_t ws_size,
                              hipStream_t stream) {
  (void)in_sizes; (void)n_in; (void)out_size; (void)ws_size;
  const float* x      = (const float*)d_in[0];
  const int*   y      = (const int*)  d_in[1];
  const float* alpha0 = (const float*)d_in[2];
  const float* alpha  = (const float*)d_in[3];
  const float* beta0  = (const float*)d_in[4];
  const float* beta   = (const float*)d_in[5];
  float* out = (float*)d_out;

  float* ws    = (float*)d_ws;
  float* linpi = ws;                                  // 2048*128 f32
  float* pipos = linpi + (size_t)N_PTS * M_DIM;       // 2048*128 f32
  float* tabp  = pipos + (size_t)N_PTS * M_DIM;       // 128*100 f32
  float* tabn  = tabp + (size_t)M_DIM * C_CLS;        // 128*100 f32
  uint16_t* xbf = (uint16_t*)(tabn + (size_t)M_DIM * C_CLS);   // 2048*1024 bf16
  uint16_t* abf = xbf + (size_t)N_PTS * F_DIM;                 // 128*1024 bf16

  k_zero<<<1, 1, 0, stream>>>(out);
  k_tabs<<<(M_DIM * C_CLS + 255) / 256, 256, 0, stream>>>(beta, tabp, tabn);
  k_cvt<<<(N_PTS * F_DIM / 4 + 255) / 256, 256, 0, stream>>>(x, xbf, N_PTS * F_DIM / 4);
  k_cvt<<<(M_DIM * F_DIM / 4 + 255) / 256, 256, 0, stream>>>(alpha, abf, M_DIM * F_DIM / 4);
  k_linear_pi<<<N_PTS / 128, 256, 0, stream>>>(xbf, abf, alpha0, linpi, pipos);
  k_gibbs<<<N_PTS / NPB, 128, 0, stream>>>(pipos, linpi, beta0, beta, y, tabp, tabn, out);
}
